// VisionTransformerMoCo_33595234189787
// MI455X (gfx1250) — matrix operand order repaired and checked by reading, not compiled, not run on hardware
//
#include <hip/hip_runtime.h>
#include <hip/hip_bf16.h>

// ---------------- model constants ----------------
#define IMG   384
#define PP    2
#define INC   3
#define CDIM  768          // embed dim
#define NPAT  192          // patches
#define NTOK  193          // tokens incl cls
#define LNUM  6
#define HNUM  12
#define HD    64
#define ENUM  8
#define BSZ   32
#define BT    (BSZ*NTOK)   // 6176 token rows
#define PK    (INC*IMG*PP) // 2304 patch features
#define C3    (3*CDIM)     // 2304

typedef __attribute__((ext_vector_type(16))) _Float16 v16h;
typedef __attribute__((ext_vector_type(8)))  _Float16 v8h;
typedef __attribute__((ext_vector_type(8)))  float    v8f;
typedef __attribute__((ext_vector_type(4)))  int      v4i;

// ---------------- async LDS staging (gfx1250 ASYNCcnt path) ----------------
#if defined(__HIP_DEVICE_COMPILE__) && __has_builtin(__builtin_amdgcn_global_load_async_to_lds_b128)
#define USE_ASYNC_LDS 1
#endif

#define AS1 __attribute__((address_space(1)))
#define AS3 __attribute__((address_space(3)))

// copy 16 bytes global -> LDS
__device__ __forceinline__ void lds_copy16(const _Float16* src, _Float16* dst) {
#ifdef USE_ASYNC_LDS
    __builtin_amdgcn_global_load_async_to_lds_b128(
        (AS1 v4i*)(uintptr_t)src,
        (AS3 v4i*)(unsigned)(uintptr_t)dst, 0, 0);
#else
    *(v8h*)dst = *(const v8h*)src;
#endif
}

__device__ __forceinline__ void lds_copy_wait() {
#ifdef USE_ASYNC_LDS
#if __has_builtin(__builtin_amdgcn_s_wait_asynccnt)
    __builtin_amdgcn_s_wait_asynccnt(0);
#else
    asm volatile("s_wait_asynccnt 0x0" ::: "memory");
#endif
#endif
}

// =======================================================================
// WMMA GEMM:  C[M,N](f32) (+)= A[M,K](f16,row-major) * W[N,K]^T (f16)
// Block = 8 waves = 128x64 output tile; B chunk (64 cols x 32 K) staged in
// LDS (double-buffered, async copy) and shared by all 8 waves; each wave
// computes 16x64 via 4 accumulators reusing one A fragment.
// MODE 0: C = A*W^T + bias
// MODE 1: C += A*W^T + bias              (single writer, residual add)
// MODE 2: atomicAdd(C, rowscale * (A*W^T + bias)); gridDim.y = expert
// =======================================================================
template<int MODE>
__global__ __launch_bounds__(256)
void gemm_wmma(const _Float16* __restrict__ A, int lda,
               const _Float16* __restrict__ Wt, int ldb, size_t eWstride,
               const float* __restrict__ bias, size_t eBstride,
               float* __restrict__ Co, int ldc,
               int tilesM, int tilesN64, int Kd,
               const float* __restrict__ scale, int sstride)
{
    // 64 cols x 32 K halves, rows padded to 40 halves (80B) => conflict-free
    __shared__ _Float16 Bs[2][64][40];

    int tid  = threadIdx.x;
    int wave = tid >> 5, lane = tid & 31;
    int hi = lane >> 4, lm = lane & 15;

    int nIdx = blockIdx.x % tilesN64;
    int mIdx = blockIdx.x / tilesN64;
    int tmi  = mIdx * 8 + wave;
    bool valid = tmi < tilesM;            // wave-uniform
    int tmc = valid ? tmi : tilesM - 1;   // clamp: stay in-bounds, skip store
    int tm = tmc << 4, tn = nIdx << 6;

    const _Float16* W = Wt + (size_t)blockIdx.y * eWstride;
    const float* bptr = bias + (size_t)blockIdx.y * eBstride;

    float sc[8];
    if (MODE == 2) {
        const float* sp = scale + blockIdx.y;
        #pragma unroll
        for (int r = 0; r < 8; ++r)
            sc[r] = sp[(size_t)(tm + r + hi*8) * sstride];
    }

    // each of the 256 threads stages one 16B segment of the 4KB B chunk
    int scol = tid >> 2, sseg = tid & 3;
    const _Float16* srcbase = W + (size_t)(tn + scol) * ldb + sseg*8;
    _Float16* dst0 = &Bs[0][scol][sseg*8];
    _Float16* dst1 = &Bs[1][scol][sseg*8];

    lds_copy16(srcbase, dst0);            // prefetch chunk 0

    v8f acc[4] = {};
    const _Float16* arow = A + (size_t)(tm + lm) * lda;
    int nk = Kd >> 5;

    for (int kt = 0; kt < nk; ++kt) {
        int buf = kt & 1;
        lds_copy_wait();
        __syncthreads();
        if (kt + 1 < nk)
            lds_copy16(srcbase + (kt+1)*32, buf ? dst0 : dst1);

        int tk = kt << 5;
        // A-frag 16x32 from global: lane half selects K {0-7,16-23}/{8-15,24-31}
        const _Float16* ap = arow + tk + hi*8;
        v8h a0 = *(const v8h*)(ap);
        v8h a1 = *(const v8h*)(ap + 16);
        v16h afrag = __builtin_shufflevector(a0, a1,
                        0,1,2,3,4,5,6,7,8,9,10,11,12,13,14,15);
        #pragma unroll
        for (int j = 0; j < 4; ++j) {
            // B-frag 32x16 from LDS: col = 16j+lm, K {0-7,16-23}/{8-15,24-31} by lane half
            const _Float16* bp = &Bs[buf][j*16 + lm][hi*8];
            v8h b0 = *(const v8h*)(bp);
            v8h b1 = *(const v8h*)(bp + 16);
            v16h bfrag = __builtin_shufflevector(b0, b1,
                            0,1,2,3,4,5,6,7,8,9,10,11,12,13,14,15);
            acc[j] = __builtin_amdgcn_wmma_f32_16x16x32_f16(
                         false, afrag, false, bfrag, (short)0, acc[j], false, false);
        }
    }

    if (!valid) return;                   // clamped waves: no store
    #pragma unroll
    for (int j = 0; j < 4; ++j) {
        int n = tn + j*16 + lm;
        float bv = bptr[n];
        #pragma unroll
        for (int r = 0; r < 8; ++r) {
            int row = tm + r + hi*8;
            float v = acc[j][r] + bv;
            float* cp = Co + (size_t)row * ldc + n;
            if (MODE == 0)      *cp = v;
            else if (MODE == 1) *cp += v;
            else if (sc[r] != 0.f) atomicAdd(cp, sc[r] * v);
        }
    }
}

// ---------------- elementwise helpers ----------------
__global__ __launch_bounds__(256)
void f32_to_f16_kernel(const float* __restrict__ s, _Float16* __restrict__ d, int n) {
    int i = blockIdx.x * 256 + threadIdx.x;
    if (i < n) d[i] = (_Float16)s[i];
}

// gather x[B,3,384,384] -> patch rows xp[B*192, 2304] (f16), read-coalesced
__global__ __launch_bounds__(256)
void patch_gather_kernel(const float* __restrict__ x, _Float16* __restrict__ xp) {
    int idx = blockIdx.x * 256 + threadIdx.x;       // over B*3*384*384
    int w  = idx % IMG;
    int t  = idx / IMG;
    int r  = t % IMG;  t /= IMG;
    int ic = t % INC;
    int b  = t / INC;
    int n = w >> 1, p = w & 1;
    int col = (ic*IMG + r)*PP + p;
    xp[(size_t)(b*NPAT + n) * PK + col] = (_Float16)x[idx];
}

// h[b,n,c] = (n==0 ? cls[c] : tok[b*192+n-1,c]) + pos[n,c]
__global__ __launch_bounds__(256)
void assemble_kernel(const float* __restrict__ tok, const float* __restrict__ cls,
                     const float* __restrict__ pos, float* __restrict__ h) {
    int idx = blockIdx.x * 256 + threadIdx.x;       // over BT*CDIM
    int c = idx % CDIM;
    int row = idx / CDIM;
    int n = row % NTOK, b = row / NTOK;
    float v = pos[n*CDIM + c];
    v += (n == 0) ? cls[c] : tok[((size_t)b*NPAT + n - 1)*CDIM + c];
    h[idx] = v;
}

// ---------------- LayerNorm (wave per token, C=768) ----------------
__global__ __launch_bounds__(256)
void ln_f16_kernel(const float* __restrict__ X, const float* __restrict__ w,
                   const float* __restrict__ b, _Float16* __restrict__ Y, int ntok)
{
    int t = blockIdx.x * 8 + (threadIdx.x >> 5);
    if (t >= ntok) return;
    int lane = threadIdx.x & 31;
    const float* x = X + (size_t)t * CDIM;
    float v[24]; float s = 0.f;
    #pragma unroll
    for (int i = 0; i < 24; ++i) { v[i] = x[lane + 32*i]; s += v[i]; }
    #pragma unroll
    for (int m = 16; m >= 1; m >>= 1) s += __shfl_xor(s, m, 32);
    float mean = s * (1.f/CDIM);
    float q = 0.f;
    #pragma unroll
    for (int i = 0; i < 24; ++i) { float d = v[i]-mean; q += d*d; }
    #pragma unroll
    for (int m = 16; m >= 1; m >>= 1) q += __shfl_xor(q, m, 32);
    float rs = rsqrtf(q * (1.f/CDIM) + 1e-5f);
    _Float16* y = Y + (size_t)t * CDIM;
    #pragma unroll
    for (int i = 0; i < 24; ++i) {
        int c = lane + 32*i;
        y[c] = (_Float16)((v[i]-mean)*rs*w[c] + b[c]);
    }
}

// final LN on cls token only -> d_out[b, colOff..colOff+767]
__global__ __launch_bounds__(256)
void ln_cls_kernel(const float* __restrict__ X, const float* __restrict__ w,
                   const float* __restrict__ b, float* __restrict__ out, int colOff)
{
    int bi = blockIdx.x * 8 + (threadIdx.x >> 5);
    if (bi >= BSZ) return;
    int lane = threadIdx.x & 31;
    const float* x = X + (size_t)bi * NTOK * CDIM;  // token 0
    float v[24]; float s = 0.f;
    #pragma unroll
    for (int i = 0; i < 24; ++i) { v[i] = x[lane + 32*i]; s += v[i]; }
    #pragma unroll
    for (int m = 16; m >= 1; m >>= 1) s += __shfl_xor(s, m, 32);
    float mean = s * (1.f/CDIM);
    float q = 0.f;
    #pragma unroll
    for (int i = 0; i < 24; ++i) { float d = v[i]-mean; q += d*d; }
    #pragma unroll
    for (int m = 16; m >= 1; m >>= 1) q += __shfl_xor(q, m, 32);
    float rs = rsqrtf(q * (1.f/CDIM) + 1e-5f);
    float* o = out + (size_t)bi * (2*CDIM) + colOff;
    #pragma unroll
    for (int i = 0; i < 24; ++i) {
        int c = lane + 32*i;
        o[c] = (v[i]-mean)*rs*w[c] + b[c];
    }
}

// ---------------- attention: one wave per (b, head, q) ----------------
__global__ __launch_bounds__(256)
void attention_kernel(const float* __restrict__ qkv, _Float16* __restrict__ o16)
{
    const int TOT = BSZ*HNUM*NTOK;
    int wid = blockIdx.x * 8 + (threadIdx.x >> 5);
    if (wid >= TOT) return;
    int lane = threadIdx.x & 31;
    int q = wid % NTOK;
    int r = wid / NTOK;
    int head = r % HNUM, b = r / HNUM;
    const float* base = qkv + (size_t)b * NTOK * C3;
    const float* qp = base + (size_t)q * C3 + head*HD;
    float qa = qp[lane] * 0.125f, qb = qp[lane+32] * 0.125f;   // HD^-0.5

    float sarr[7];
    #pragma unroll
    for (int i = 0; i < 7; ++i) sarr[i] = -1e30f;
    #pragma unroll
    for (int sblk = 0; sblk < 7; ++sblk) {
        int jmax = (sblk == 6) ? 1 : 32;
        for (int jj = 0; jj < jmax; ++jj) {
            int j = sblk*32 + jj;
            const float* kp = base + (size_t)j * C3 + CDIM + head*HD;
            float part = qa*kp[lane] + qb*kp[lane+32];
            #pragma unroll
            for (int m = 16; m >= 1; m >>= 1) part += __shfl_xor(part, m, 32);
            if (jj == lane) sarr[sblk] = part;
        }
    }
    float mx = -1e30f;
    #pragma unroll
    for (int i = 0; i < 7; ++i) mx = fmaxf(mx, sarr[i]);
    #pragma unroll
    for (int m = 16; m >= 1; m >>= 1) mx = fmaxf(mx, __shfl_xor(mx, m, 32));
    float parr[7]; float sum = 0.f;
    #pragma unroll
    for (int i = 0; i < 7; ++i) { parr[i] = expf(sarr[i]-mx); sum += parr[i]; }
    #pragma unroll
    for (int m = 16; m >= 1; m >>= 1) sum += __shfl_xor(sum, m, 32);
    float inv = 1.f / sum;

    float a0 = 0.f, a1 = 0.f;
    #pragma unroll
    for (int sblk = 0; sblk < 7; ++sblk) {
        int jmax = (sblk == 6) ? 1 : 32;
        for (int jj = 0; jj < jmax; ++jj) {
            int j = sblk*32 + jj;
            float pj = __shfl(parr[sblk], jj, 32);
            const float* vp = base + (size_t)j * C3 + 2*CDIM + head*HD;
            a0 += pj * vp[lane];
            a1 += pj * vp[lane+32];
        }
    }
    _Float16* op = o16 + ((size_t)(b*NTOK + q)) * CDIM + head*HD;
    op[lane]    = (_Float16)(a0 * inv);
    op[lane+32] = (_Float16)(a1 * inv);
}

// ---------------- threefry2x32-20 (JAX PRNG) ----------------
__device__ inline unsigned rotl32(unsigned x, int n) { return (x<<n) | (x>>(32-n)); }

__device__ inline void tf2x32(unsigned k0, unsigned k1, unsigned x0, unsigned x1,
                              unsigned& o0, unsigned& o1)
{
    const unsigned R[8] = {13,15,26,6,17,29,16,24};
    unsigned ks0 = k0, ks1 = k1, ks2 = 0x1BD11BDAu ^ k0 ^ k1;
    x0 += ks0; x1 += ks1;
    unsigned ks[3] = {ks0, ks1, ks2};
    #pragma unroll
    for (int i = 0; i < 5; ++i) {
        #pragma unroll
        for (int r = 0; r < 4; ++r) {
            x0 += x1; x1 = rotl32(x1, R[(i & 1)*4 + r]); x1 ^= x0;
        }
        x0 += ks[(i+1)%3];
        x1 += ks[(i+2)%3] + (unsigned)(i+1);
    }
    o0 = x0; o1 = x1;
}

__device__ inline float jax_noise(unsigned k0, unsigned k1, int i)
{
    const int half = (BT * ENUM) / 2;              // 24704
    int j = (i < half) ? i : i - half;
    unsigned o0, o1;
    tf2x32(k0, k1, (unsigned)j, (unsigned)(j + half), o0, o1);
    unsigned bits = (i < half) ? o0 : o1;
    float f = __uint_as_float(0x3f800000u | (bits >> 9)) - 1.0f;   // [0,1)
    const float lo = -0.99999994f;
    float u = lo + f * (1.0f - lo);
    return 1.41421356f * erfinvf(u) * (1.0f/ENUM);
}

// ---------------- router: wave per token ----------------
__global__ __launch_bounds__(256)
void router_kernel(const _Float16* __restrict__ y16, const float* __restrict__ rw,
                   const float* __restrict__ rb, const float* __restrict__ rlw,
                   const float* __restrict__ rlb, float* __restrict__ wdense,
                   int view, int layer)
{
    int t = blockIdx.x * 8 + (threadIdx.x >> 5);
    if (t >= BT) return;
    int lane = threadIdx.x & 31;
    const _Float16* yp = y16 + (size_t)t * CDIM;
    float yv[24];
    #pragma unroll
    for (int i = 0; i < 24; ++i) yv[i] = (float)yp[lane + 32*i];

    float logit[8];
    #pragma unroll
    for (int e = 0; e < ENUM; ++e) {
        const float* wp = rw + (size_t)e * CDIM;
        float p = 0.f;
        #pragma unroll
        for (int i = 0; i < 24; ++i) p += yv[i] * wp[lane + 32*i];
        #pragma unroll
        for (int m = 16; m >= 1; m >>= 1) p += __shfl_xor(p, m, 32);
        logit[e] = p + rb[e];
    }
    float mean = 0.f;
    #pragma unroll
    for (int e = 0; e < ENUM; ++e) mean += logit[e];
    mean *= (1.f/ENUM);
    float var = 0.f;
    #pragma unroll
    for (int e = 0; e < ENUM; ++e) { float d = logit[e]-mean; var += d*d; }
    float rs = rsqrtf(var * (1.f/ENUM) + 1e-5f);
    #pragma unroll
    for (int e = 0; e < ENUM; ++e) logit[e] = (logit[e]-mean)*rs*rlw[e] + rlb[e];
    float mx = logit[0];
    #pragma unroll
    for (int e = 1; e < ENUM; ++e) mx = fmaxf(mx, logit[e]);
    float se = 0.f; float rwts[8];
    #pragma unroll
    for (int e = 0; e < ENUM; ++e) { rwts[e] = expf(logit[e]-mx); se += rwts[e]; }
    float inv = 1.f/se;
    unsigned b0, b1, k0, k1;
    tf2x32(0u, 1234u, 0u, (unsigned)view,  b0, b1);   // fold_in(key(1234), view)
    tf2x32(b0, b1,  0u, (unsigned)layer, k0, k1);     // fold_in(base, layer)
    #pragma unroll
    for (int e = 0; e < ENUM; ++e) rwts[e] = rwts[e]*inv + jax_noise(k0, k1, t*ENUM + e);
    int i1 = 0; float v1 = rwts[0];
    #pragma unroll
    for (int e = 1; e < ENUM; ++e) if (rwts[e] > v1) { v1 = rwts[e]; i1 = e; }
    int i2 = -1; float v2 = -1e30f;
    #pragma unroll
    for (int e = 0; e < ENUM; ++e) if (e != i1 && rwts[e] > v2) { v2 = rwts[e]; i2 = e; }
    float w1 = 1.f / (1.f + expf(v2 - v1));           // softmax([v1,v2])
    float w2 = 1.f - w1;
    if (lane < ENUM) {
        float* wd = wdense + (size_t)t * ENUM;
        wd[lane] = (lane == i1) ? w1 : (lane == i2) ? w2 : 0.f;
    }
}

// =======================================================================
extern "C" void kernel_launch(void* const* d_in, const int* in_sizes, int n_in,
                              void* d_out, int out_size, void* d_ws, size_t ws_size,
                              hipStream_t stream)
{
    const float* x1      = (const float*)d_in[0];
    const float* x2      = (const float*)d_in[1];
    const float* patch_w = (const float*)d_in[2];
    const float* patch_b = (const float*)d_in[3];
    const float* cls_tok = (const float*)d_in[4];
    const float* ln1_w   = (const float*)d_in[5];
    const float* ln1_b   = (const float*)d_in[6];
    const float* qkv_w   = (const float*)d_in[7];
    const float* qkv_b   = (const float*)d_in[8];
    const float* proj_w  = (const float*)d_in[9];
    const float* proj_b  = (const float*)d_in[10];
    const float* ln2_w   = (const float*)d_in[11];
    const float* ln2_b   = (const float*)d_in[12];
    const float* route_w = (const float*)d_in[13];
    const float* route_b = (const float*)d_in[14];
    const float* rln_w   = (const float*)d_in[15];
    const float* rln_b   = (const float*)d_in[16];
    const float* exp_w   = (const float*)d_in[17];
    const float* exp_b   = (const float*)d_in[18];
    const float* norm_w  = (const float*)d_in[19];
    const float* norm_b  = (const float*)d_in[20];
    const float* pos_emb = (const float*)d_in[21];
    float* out = (float*)d_out;

    // ---- workspace carve (≈133 MB) ----
    size_t off = 0;
    auto carve = [&](size_t bytes) {
        void* p = (char*)d_ws + off;
        off += (bytes + 255) & ~(size_t)255;
        return p;
    };
    float*    h      = (float*)   carve((size_t)BT * CDIM * 4);
    float*    qkvbuf = (float*)   carve((size_t)BT * C3   * 4);  // also patch-tok staging
    _Float16* y16    = (_Float16*)carve((size_t)BT * CDIM * 2);
    _Float16* o16    = (_Float16*)carve((size_t)BT * CDIM * 2);
    _Float16* xp16   = (_Float16*)carve((size_t)BSZ*NPAT * PK * 2);
    _Float16* wbuf   = (_Float16*)carve((size_t)ENUM * CDIM * CDIM * 2); // 9.4MB reuse
    float*    wdense = (float*)   carve((size_t)BT * ENUM * 4);

    const int LNBLK = BT / 8;                        // 772
    const int ATBLK = (BSZ*HNUM*NTOK) / 8;           // 9264
    auto gemmGrid = [](int tilesM, int tilesN64) {
        return (unsigned)(((tilesM + 7) / 8) * tilesN64);
    };

    for (int view = 0; view < 2; ++view) {
        const float* xin = view ? x2 : x1;

        // ---- patch embed ----
        f32_to_f16_kernel<<<(CDIM*PK + 255)/256, 256, 0, stream>>>(patch_w, wbuf, CDIM*PK);
        patch_gather_kernel<<<(BSZ*INC*IMG*IMG)/256, 256, 0, stream>>>(xin, xp16);
        gemm_wmma<0><<<dim3(gemmGrid(BSZ*NPAT/16, CDIM/64), 1), 256, 0, stream>>>(
            xp16, PK, wbuf, PK, 0, patch_b, 0, qkvbuf, CDIM,
            BSZ*NPAT/16, CDIM/64, PK, nullptr, 0);
        assemble_kernel<<<(BT*CDIM)/256, 256, 0, stream>>>(qkvbuf, cls_tok, pos_emb, h);

        for (int l = 0; l < LNUM; ++l) {
            // ---- attention ----
            ln_f16_kernel<<<LNBLK, 256, 0, stream>>>(h, ln1_w + l*CDIM, ln1_b + l*CDIM, y16, BT);
            f32_to_f16_kernel<<<(C3*CDIM + 255)/256, 256, 0, stream>>>(
                qkv_w + (size_t)l*C3*CDIM, wbuf, C3*CDIM);
            gemm_wmma<0><<<dim3(gemmGrid(BT/16, C3/64), 1), 256, 0, stream>>>(
                y16, CDIM, wbuf, CDIM, 0, qkv_b + l*C3, 0, qkvbuf, C3,
                BT/16, C3/64, CDIM, nullptr, 0);
            attention_kernel<<<ATBLK, 256, 0, stream>>>(qkvbuf, o16);
            f32_to_f16_kernel<<<(CDIM*CDIM + 255)/256, 256, 0, stream>>>(
                proj_w + (size_t)l*CDIM*CDIM, wbuf, CDIM*CDIM);
            gemm_wmma<1><<<dim3(gemmGrid(BT/16, CDIM/64), 1), 256, 0, stream>>>(
                o16, CDIM, wbuf, CDIM, 0, proj_b + l*CDIM, 0, h, CDIM,
                BT/16, CDIM/64, CDIM, nullptr, 0);

            // ---- MoE ----
            ln_f16_kernel<<<LNBLK, 256, 0, stream>>>(h, ln2_w + l*CDIM, ln2_b + l*CDIM, y16, BT);
            router_kernel<<<LNBLK, 256, 0, stream>>>(
                y16, route_w + (size_t)l*ENUM*CDIM, route_b + l*ENUM,
                rln_w + l*ENUM, rln_b + l*ENUM, wdense, view, l);
            f32_to_f16_kernel<<<(ENUM*CDIM*CDIM + 255)/256, 256, 0, stream>>>(
                exp_w + (size_t)l*ENUM*CDIM*CDIM, wbuf, ENUM*CDIM*CDIM);
            gemm_wmma<2><<<dim3(gemmGrid(BT/16, CDIM/64), ENUM), 256, 0, stream>>>(
                y16, CDIM, wbuf, CDIM, (size_t)CDIM*CDIM,
                exp_b + (size_t)l*ENUM*CDIM, CDIM, h, CDIM,
                BT/16, CDIM/64, CDIM, wdense, ENUM);
        }
        // ---- final LN on cls -> output half ----
        ln_cls_kernel<<<4, 256, 0, stream>>>(h, norm_w, norm_b, out, view*CDIM);
    }
    (void)in_sizes; (void)n_in; (void)out_size; (void)ws_size;
}
